// MultiHeadAttention_24850680775204
// MI455X (gfx1250) — matrix operand order repaired and checked by reading, not compiled, not run on hardware
//
#include <hip/hip_runtime.h>
#include <hip/hip_bf16.h>

// MI455X / gfx1250, wave32. BF16 WMMA + TDM (tensor_load_to_lds) B-tile staging.

typedef __attribute__((ext_vector_type(16))) __bf16       v16bf;
typedef __attribute__((ext_vector_type(8)))  float        v8f;
typedef __attribute__((ext_vector_type(4)))  unsigned int u32x4;
typedef __attribute__((ext_vector_type(8)))  int          i32x8;
typedef __attribute__((ext_vector_type(4)))  int          i32x4;

#define D_MODEL 1024
#define NHEADS  16
#define DK      64
#define SEQ     2048
#define BATCH   2
#define NEG_BIG (-3.0e38f)
#define KSTEPS  (D_MODEL / 32)

// ---------- WMMA wrapper ----------
__device__ __forceinline__ v8f wmma_bf16(v16bf a, v16bf b, v8f c) {
    return __builtin_amdgcn_wmma_f32_16x16x32_bf16(false, a, false, b,
                                                   (short)0, c, false, false);
}

// ---------- A-fragment builders (16x32 bf16, ISA layout) ----------
// lane L: m = L%16, half = L/16; elem j<8 -> K=half*8+j ; j>=8 -> K=16+half*8+(j-8)
__device__ __forceinline__ v16bf a_frag_bf16(const __bf16* row_k0, int half) {
    v16bf a;
    const __bf16* p0 = row_k0 + half * 8;
    const __bf16* p1 = row_k0 + 16 + half * 8;
#pragma unroll
    for (int t = 0; t < 8; ++t) { a[t] = p0[t]; a[t + 8] = p1[t]; }
    return a;
}
__device__ __forceinline__ v16bf a_frag_f32(const float* row_k0, int half) {
    v16bf a;
    const float* p0 = row_k0 + half * 8;
    const float* p1 = row_k0 + 16 + half * 8;
#pragma unroll
    for (int t = 0; t < 8; ++t) { a[t] = (__bf16)p0[t]; a[t + 8] = (__bf16)p1[t]; }
    return a;
}
// B-fragment (32x16 bf16): lane L: n = L%16, half = L/16; elem j<8 -> K=half*8+j ; j>=8 -> K=16+half*8+(j-8)
__device__ __forceinline__ v16bf b_frag(const __bf16* p, int ld, int half) {
    v16bf b;
#pragma unroll
    for (int t = 0; t < 8; ++t) { b[t] = p[(size_t)(half * 8 + t) * ld]; b[t + 8] = p[(size_t)(16 + half * 8 + t) * ld]; }
    return b;
}

// ---------- TDM: 2D tile (rows=32, cols=64 bf16) global -> LDS ----------
// D# per CDNA5 ISA 8.3/8.4: group0 = {flags, lds_addr, gaddr_lo, gaddr_hi|type},
// group1 packs data_size / tensor dims / tile dims / strides.
// This toolchain exposes the 6-arg builtin (g0, g1, g2, g3, g4, cpol).
__device__ __forceinline__ void tdm_load_tile_32x64(unsigned lds_off, const __bf16* gsrc) {
    size_t ga = (size_t)gsrc;
    u32x4 g0;
    g0[0] = 1u;                                           // count=1, user descriptor
    g0[1] = lds_off;                                      // LDS byte address
    g0[2] = (unsigned)(ga & 0xFFFFFFFFu);                 // global_addr[31:0]
    g0[3] = (unsigned)((ga >> 32) & 0x01FFFFFFu) | (2u << 30);  // addr[56:32] | type=2
    i32x8 g1;
    g1[0] = (1 << 16);                 // wg_mask=0, data_size=1 (2 bytes), no flags
    g1[1] = (int)((unsigned)D_MODEL << 16);   // abar=0 | tensor_dim0[15:0]=1024
    g1[2] = (int)((unsigned)D_MODEL << 16);   // tensor_dim0[31:16]=0 | tensor_dim1[15:0]=1024
    g1[3] = (int)(64u << 16);                 // tensor_dim1[31:16]=0 | tile_dim0=64
    g1[4] = 32;                               // tile_dim1=32 | tile_dim2=0
    g1[5] = D_MODEL;                          // tensor_dim0_stride[31:0]=1024
    g1[6] = 0;                                // stride0[47:32]=0 | stride1[15:0]=0
    g1[7] = 0;
    i32x4 gz4 = {0, 0, 0, 0};
    i32x8 gz8 = {0, 0, 0, 0, 0, 0, 0, 0};
    __builtin_amdgcn_tensor_load_to_lds(g0, g1, gz4, gz4, gz8, 0);
}

// ---------- Kernel 1: weight cast + transpose (fp32 [n][k] -> bf16 [k][n]) ----------
__global__ void mha_wt_cast(const float* __restrict__ W, __bf16* __restrict__ Wt) {
    int i = blockIdx.x * 256 + threadIdx.x;   // i = k*D + n
    int k = i / D_MODEL, n = i % D_MODEL;
    Wt[i] = (__bf16)W[(size_t)n * D_MODEL + k];
}

// ---------- Kernel 2: projection GEMM  Y = X(fp32) @ Wt(bf16) ----------
// Block: 128 rows x 64 cols (8 waves x 16 rows, each wave all 64 cols).
// B tile (32x64) staged by TDM into double-buffered LDS; one head per 64-col tile.
// mode 0/2: Y[b][h][s][d] ; mode 1: Y[b][h][d][s]
__global__ void mha_proj_gemm(const float* __restrict__ X, const __bf16* __restrict__ Wt,
                              __bf16* __restrict__ Y, int mode) {
    __shared__ __bf16 Bsm[2][32][64];                    // 8 KB
    const int lane = threadIdx.x & 31, half = lane >> 4, l16 = lane & 15;
    const int w = threadIdx.x >> 5;
    const int blockRow = blockIdx.x >> 4;                // 32 row tiles of 128
    const int hblk = blockIdx.x & 15;                    // 16 col tiles of 64 = head id
    const int m0 = blockRow * 128 + w * 16;
    const int n0 = hblk * 64;

    const float* arow = X + (size_t)(m0 + l16) * D_MODEL;
    v8f acc[4] = {};

    if (w == 0)  // prime buffer 0
        tdm_load_tile_32x64((unsigned)(size_t)&Bsm[0][0][0], Wt + (size_t)0 * D_MODEL + n0);

    for (int i = 0; i < KSTEPS; ++i) {
        const int kb = i * 32, buf = i & 1;
        if (w == 0) {
            if (i + 1 < KSTEPS) {
                tdm_load_tile_32x64((unsigned)(size_t)&Bsm[buf ^ 1][0][0],
                                    Wt + (size_t)(kb + 32) * D_MODEL + n0);
                __builtin_amdgcn_s_wait_tensorcnt(1);    // tile for THIS iter is done
            } else {
                __builtin_amdgcn_s_wait_tensorcnt(0);
            }
        }
        __syncthreads();                                 // tile visible to all waves
        v16bf a = a_frag_f32(arow + kb, half);
        const __bf16* brow = &Bsm[buf][0][l16];
#pragma unroll
        for (int c = 0; c < 4; ++c)
            acc[c] = wmma_bf16(a, b_frag(brow + c * 16, 64, half), acc[c]);
        __syncthreads();                                 // done reading before overwrite
    }
#pragma unroll
    for (int c = 0; c < 4; ++c)
#pragma unroll
        for (int r = 0; r < 8; ++r) {
            int m = m0 + half * 8 + r;
            int bb = m >> 11, s = m & (SEQ - 1);
            int d = c * 16 + l16;
            size_t idx = (mode == 1)
                       ? ((size_t)(bb * NHEADS + hblk) * DK + d) * SEQ + s
                       : ((size_t)(bb * NHEADS + hblk) * SEQ + s) * DK + d;
            Y[idx] = (__bf16)acc[c][r];
        }
}

// ---------- Kernel 3: causal flash attention, one wave per (b,h,16-query tile) ----------
__global__ void mha_flash(const __bf16* __restrict__ Qh, const __bf16* __restrict__ Kt,
                          const __bf16* __restrict__ Vh, __bf16* __restrict__ AO) {
    __shared__ __bf16 Pl[8][16][32];          // per-wave P staging (C-layout -> A-layout)
    const int w = threadIdx.x >> 5, lane = threadIdx.x & 31;
    const int half = lane >> 4, l16 = lane & 15;
    const int tile = blockIdx.x * 8 + w;      // 4096 tiles exactly
    const int qtpb = SEQ / 16;
    const int b = tile / (NHEADS * qtpb);
    const int h = (tile / qtpb) % NHEADS;
    const int qb = tile % qtpb;
    const int bh = b * NHEADS + h;

    const __bf16* qrow = Qh + ((size_t)bh * SEQ + qb * 16 + l16) * DK;
    v16bf aq0 = a_frag_bf16(qrow, half);
    v16bf aq1 = a_frag_bf16(qrow + 32, half);

    float Mr[8], Lr[8];
    v8f o[4] = {};
#pragma unroll
    for (int r = 0; r < 8; ++r) { Mr[r] = NEG_BIG; Lr[r] = 0.f; }

    const int qmax = qb * 16 + 15;
    for (int n0 = 0; n0 <= qmax; n0 += 32) {
        v8f sc0 = {}, sc1 = {};
        {
            const __bf16* kb0 = Kt + ((size_t)bh * DK) * SEQ + n0 + l16;
            sc0 = wmma_bf16(aq0, b_frag(kb0, SEQ, half), sc0);
            sc1 = wmma_bf16(aq0, b_frag(kb0 + 16, SEQ, half), sc1);
            const __bf16* kb1 = Kt + ((size_t)bh * DK + 32) * SEQ + n0 + l16;
            sc0 = wmma_bf16(aq1, b_frag(kb1, SEQ, half), sc0);
            sc1 = wmma_bf16(aq1, b_frag(kb1 + 16, SEQ, half), sc1);
        }
#pragma unroll
        for (int r = 0; r < 8; ++r) {
            const int mrow = qb * 16 + half * 8 + r;
            float v0 = sc0[r] * 0.125f;                 // 1/sqrt(64)
            float v1 = sc1[r] * 0.125f;
            if (n0 + l16 > mrow)      v0 = NEG_BIG;
            if (n0 + 16 + l16 > mrow) v1 = NEG_BIG;
            float mx = fmaxf(v0, v1);
#pragma unroll
            for (int off = 1; off < 16; off <<= 1)
                mx = fmaxf(mx, __shfl_xor(mx, off, 32));
            const float mnew = fmaxf(Mr[r], mx);
            const float scl = __expf(Mr[r] - mnew);
            const float p0 = __expf(v0 - mnew);
            const float p1 = __expf(v1 - mnew);
            float sum = p0 + p1;
#pragma unroll
            for (int off = 1; off < 16; off <<= 1)
                sum += __shfl_xor(sum, off, 32);
            Mr[r] = mnew;
            Lr[r] = Lr[r] * scl + sum;
            o[0][r] *= scl; o[1][r] *= scl; o[2][r] *= scl; o[3][r] *= scl;
            Pl[w][half * 8 + r][l16]      = (__bf16)p0;
            Pl[w][half * 8 + r][16 + l16] = (__bf16)p1;
        }
        asm volatile("s_wait_dscnt 0" ::: "memory");    // intra-wave LDS RAW
        v16bf ap = a_frag_bf16(&Pl[w][l16][0], half);
        const __bf16* vrow = Vh + ((size_t)bh * SEQ + n0) * DK + l16;
#pragma unroll
        for (int c = 0; c < 4; ++c)
            o[c] = wmma_bf16(ap, b_frag(vrow + c * 16, DK, half), o[c]);
    }
#pragma unroll
    for (int r = 0; r < 8; ++r) {
        const float inv = 1.f / Lr[r];
        const int mrow = qb * 16 + half * 8 + r;
        const size_t base = ((size_t)b * SEQ + mrow) * D_MODEL + h * DK + l16;
#pragma unroll
        for (int c = 0; c < 4; ++c)
            AO[base + c * 16] = (__bf16)(o[c][r] * inv);
    }
}

// ---------- Kernel 4: output projection  out(fp32) = AO(bf16) @ Wot(bf16) ----------
__global__ void mha_out_gemm(const __bf16* __restrict__ A, const __bf16* __restrict__ Wt,
                             float* __restrict__ C) {
    __shared__ __bf16 Bsm[2][32][64];
    const int lane = threadIdx.x & 31, half = lane >> 4, l16 = lane & 15;
    const int w = threadIdx.x >> 5;
    const int blockRow = blockIdx.x >> 4;
    const int blockCol = blockIdx.x & 15;
    const int m0 = blockRow * 128 + w * 16;
    const int n0 = blockCol * 64;

    const __bf16* arow = A + (size_t)(m0 + l16) * D_MODEL;
    v8f acc[4] = {};

    if (w == 0)
        tdm_load_tile_32x64((unsigned)(size_t)&Bsm[0][0][0], Wt + n0);

    for (int i = 0; i < KSTEPS; ++i) {
        const int kb = i * 32, buf = i & 1;
        if (w == 0) {
            if (i + 1 < KSTEPS) {
                tdm_load_tile_32x64((unsigned)(size_t)&Bsm[buf ^ 1][0][0],
                                    Wt + (size_t)(kb + 32) * D_MODEL + n0);
                __builtin_amdgcn_s_wait_tensorcnt(1);
            } else {
                __builtin_amdgcn_s_wait_tensorcnt(0);
            }
        }
        __syncthreads();
        v16bf a = a_frag_bf16(arow + kb, half);
        const __bf16* brow = &Bsm[buf][0][l16];
#pragma unroll
        for (int c = 0; c < 4; ++c)
            acc[c] = wmma_bf16(a, b_frag(brow + c * 16, 64, half), acc[c]);
        __syncthreads();
    }
#pragma unroll
    for (int c = 0; c < 4; ++c)
#pragma unroll
        for (int r = 0; r < 8; ++r)
            C[(size_t)(m0 + half * 8 + r) * D_MODEL + n0 + c * 16 + l16] = acc[c][r];
}

extern "C" void kernel_launch(void* const* d_in, const int* in_sizes, int n_in,
                              void* d_out, int out_size, void* d_ws, size_t ws_size,
                              hipStream_t stream) {
    (void)in_sizes; (void)n_in; (void)out_size; (void)ws_size;
    const float* q  = (const float*)d_in[0];
    const float* k  = (const float*)d_in[1];
    const float* v  = (const float*)d_in[2];
    const float* Wq = (const float*)d_in[3];
    const float* Wk = (const float*)d_in[4];
    const float* Wv = (const float*)d_in[5];
    const float* Wo = (const float*)d_in[6];
    float* out = (float*)d_out;

    char* ws = (char*)d_ws;
    const size_t WSZ = (size_t)D_MODEL * D_MODEL * sizeof(__bf16);        // 2 MB
    const size_t ASZ = (size_t)BATCH * SEQ * D_MODEL * sizeof(__bf16);    // 8 MB
    __bf16* Wtq = (__bf16*)(ws);
    __bf16* Wtk = (__bf16*)(ws + WSZ);
    __bf16* Wtv = (__bf16*)(ws + 2 * WSZ);
    __bf16* Wto = (__bf16*)(ws + 3 * WSZ);
    __bf16* Qh  = (__bf16*)(ws + 4 * WSZ);
    __bf16* Kt  = (__bf16*)(ws + 4 * WSZ + ASZ);
    __bf16* Vh  = (__bf16*)(ws + 4 * WSZ + 2 * ASZ);
    __bf16* AO  = (__bf16*)(ws + 4 * WSZ + 3 * ASZ);

    const int castBlocks = (D_MODEL * D_MODEL) / 256;                     // 4096
    mha_wt_cast<<<castBlocks, 256, 0, stream>>>(Wq, Wtq);
    mha_wt_cast<<<castBlocks, 256, 0, stream>>>(Wk, Wtk);
    mha_wt_cast<<<castBlocks, 256, 0, stream>>>(Wv, Wtv);
    mha_wt_cast<<<castBlocks, 256, 0, stream>>>(Wo, Wto);

    const int gemmBlocks = (BATCH * SEQ / 128) * (D_MODEL / 64);          // 512, exact
    mha_proj_gemm<<<gemmBlocks, 256, 0, stream>>>(q, Wtq, Qh, 0);
    mha_proj_gemm<<<gemmBlocks, 256, 0, stream>>>(k, Wtk, Kt, 1);
    mha_proj_gemm<<<gemmBlocks, 256, 0, stream>>>(v, Wtv, Vh, 2);

    const int attnBlocks = BATCH * NHEADS * (SEQ / 16) / 8;               // 512, exact
    mha_flash<<<attnBlocks, 256, 0, stream>>>(Qh, Kt, Vh, AO);

    mha_out_gemm<<<gemmBlocks, 256, 0, stream>>>(AO, Wto, out);
}
